// ProteinMPNN_80839874445559
// MI455X (gfx1250) — matrix operand order repaired and checked by reading, not compiled, not run on hardware
//
#include <hip/hip_runtime.h>
#include <hip/hip_bf16.h>
#include <math.h>

typedef __attribute__((ext_vector_type(16))) _Float16 v16h;
typedef __attribute__((ext_vector_type(8)))  _Float16 v8h;
typedef __attribute__((ext_vector_type(8)))  float    v8f;

#define Bn 4
#define Ln 2048
#define Kn 48
#define EDIM 128
#define EIN 416          // 16 pos + 25*16 rbf
#define FSTR 424         // padded feats row stride (halfs); 848B = 16B multiple
#define NPAIR 25
#define MAXREL 32
#define NCHUNK 13        // 416 / 32
#define NTILES 8         // 128 / 16
#define MT 3             // 48 edges / 16 rows per M-tile

// atom order in atoms5: N=0, Ca=1, C=2, O=3, Cb=4
__device__ __constant__ int c_PA[NPAIR] = {1,0,2,3,4,1,1,1,1,0,0,0,4,4,3,0,2,3,4,2,3,4,2,3,2};
__device__ __constant__ int c_PB[NPAIR] = {1,0,2,3,4,0,2,3,4,2,3,4,2,3,2,1,1,1,1,0,0,0,4,4,3};

// ---------------- Kernel 0: build 5-atom table incl. virtual Cb ----------------
__global__ __launch_bounds__(256)
void pm_atoms_kernel(const float* __restrict__ X, float* __restrict__ atoms5) {
    int g = blockIdx.x * blockDim.x + threadIdx.x;
    if (g >= Bn * Ln) return;
    const float* x = X + (size_t)g * 12;
    float Nx=x[0],  Ny=x[1],  Nz=x[2];
    float Cax=x[3], Cay=x[4], Caz=x[5];
    float Cx=x[6],  Cy=x[7],  Cz=x[8];
    float Ox=x[9],  Oy=x[10], Oz=x[11];
    float bx=Cax-Nx, by=Cay-Ny, bz=Caz-Nz;
    float cx=Cx-Cax, cy=Cy-Cay, cz=Cz-Caz;
    float ax = by*cz - bz*cy, ay = bz*cx - bx*cz, az = bx*cy - by*cx;
    float Cbx = -0.58273431f*ax + 0.56802827f*bx - 0.54067466f*cx + Cax;
    float Cby = -0.58273431f*ay + 0.56802827f*by - 0.54067466f*cy + Cay;
    float Cbz = -0.58273431f*az + 0.56802827f*bz - 0.54067466f*cz + Caz;
    float* o = atoms5 + (size_t)g * 15;
    o[0]=Nx;  o[1]=Ny;  o[2]=Nz;
    o[3]=Cax; o[4]=Cay; o[5]=Caz;
    o[6]=Cx;  o[7]=Cy;  o[8]=Cz;
    o[9]=Ox;  o[10]=Oy; o[11]=Oz;
    o[12]=Cbx;o[13]=Cby;o[14]=Cbz;
}

// -------- Kernel 1: repack edge_W (416x128 f32) into f16 WMMA B-fragment layout --------
// layout: [kk 0..12][ntile 0..7][lane 0..31][t 0..15]
//   lane<16 : K = kk*32 + t      , col = ntile*16 + lane
//   lane>=16: K = kk*32 + 16 + t , col = ntile*16 + (lane-16)
__global__ __launch_bounds__(256)
void pm_wrepack_kernel(const float* __restrict__ W, _Float16* __restrict__ wp) {
    int g = blockIdx.x * blockDim.x + threadIdx.x;
    if (g >= NCHUNK * NTILES * 32 * 16) return;
    int t    =  g        & 15;
    int lane = (g >> 4)  & 31;
    int nt   = (g >> 9)  & 7;
    int kk   =  g >> 12;
    int K    = kk * 32 + (lane >> 4) * 16 + t;
    int col  = nt * 16 + (lane & 15);
    wp[g] = (_Float16)W[(size_t)K * EDIM + col];
}

// ---------------- Kernel 2: masked distances + top-K (register-resident) ----------------
// 256 threads; thread t owns distances for j = t, t+256, ..., t+7*256 (in registers).
__global__ __launch_bounds__(256)
void pm_topk_kernel(const float* __restrict__ X, const float* __restrict__ mask,
                    int* __restrict__ eidx, float* __restrict__ eidx_out) {
    __shared__ unsigned long long wred[8];
    __shared__ unsigned long long s_win;
    __shared__ float fred[8];
    int bi  = blockIdx.x;
    int b   = bi / Ln;
    int i   = bi - b * Ln;
    int tid = threadIdx.x;
    int lane = tid & 31, wave = tid >> 5;
    const float* Xb = X + (size_t)b * Ln * 12;
    float cx = Xb[i*12+3], cy = Xb[i*12+4], cz = Xb[i*12+5];
    float mi = mask[b*Ln + i];

    float d[8];
    float lmax = 0.f;
    #pragma unroll
    for (int r = 0; r < 8; ++r) {
        int j = tid + r * 256;
        float dx = cx - Xb[j*12+3], dy = cy - Xb[j*12+4], dz = cz - Xb[j*12+5];
        float m2 = mi * mask[b*Ln + j];
        float D = m2 * sqrtf(dx*dx + dy*dy + dz*dz + 1e-6f);
        d[r] = D;
        lmax = fmaxf(lmax, D);
    }
    #pragma unroll
    for (int o = 16; o > 0; o >>= 1) lmax = fmaxf(lmax, __shfl_xor(lmax, o, 32));
    if (lane == 0) fred[wave] = lmax;
    __syncthreads();
    float Dmax = fred[0];
    #pragma unroll
    for (int w = 1; w < 8; ++w) Dmax = fmaxf(Dmax, fred[w]);
    #pragma unroll
    for (int r = 0; r < 8; ++r) {
        int j = tid + r * 256;
        float m2 = mi * mask[b*Ln + j];
        d[r] += (1.f - m2) * Dmax;            // D_adjust
    }

    for (int k = 0; k < Kn; ++k) {
        unsigned long long best = ~0ull;
        #pragma unroll
        for (int r = 0; r < 8; ++r) {
            unsigned long long key =
                ((unsigned long long)__float_as_uint(d[r]) << 32) |
                (unsigned)(tid + r * 256);
            best = key < best ? key : best;    // min dist, tie -> lower index
        }
        #pragma unroll
        for (int o = 16; o > 0; o >>= 1) {
            unsigned long long other = __shfl_xor(best, o, 32);
            best = other < best ? other : best;
        }
        if (lane == 0) wred[wave] = best;
        __syncthreads();
        if (tid == 0) {
            unsigned long long w = wred[0];
            #pragma unroll
            for (int q = 1; q < 8; ++q) { unsigned long long o = wred[q]; w = o < w ? o : w; }
            s_win = w;
            int js = (int)(w & 0xFFFFFFFFull);
            size_t o = (size_t)bi * Kn + k;
            eidx[o] = js;
            eidx_out[o] = (float)js;
        }
        __syncthreads();
        int js = (int)(s_win & 0xFFFFFFFFull);
        if (tid == (js & 255)) d[js >> 8] = 3.0e38f;   // remove winner from owner's slot
    }
}

// ---------------- Kernel 3: features + WMMA GEMM + LayerNorm ----------------
// one block = one (b,i): all 48 edges (3 M-tiles); 256 threads = 8 waves (one N-tile each).
// B fragments loaded once per K-chunk (reused by 3 WMMAs); A fragments double-buffered so
// each chunk's ds_loads overlap the previous chunk's WMMAs (keep the XDL pipe fed).
__global__ __launch_bounds__(256)
void pm_edge_kernel(const float* __restrict__ atoms5, const int* __restrict__ eidx,
                    const int* __restrict__ ridx, const int* __restrict__ cenc,
                    const float* __restrict__ posW, const float* __restrict__ posB,
                    const _Float16* __restrict__ wp,
                    const float* __restrict__ lng, const float* __restrict__ lnb,
                    float* __restrict__ out) {
    // feats (48 x 424 f16 = 40,704B) aliased with etile (48 x 132 f32 = 25,344B):
    // feats dead after WMMA loop; barrier separates the lifetimes.
    __shared__ __align__(32) char pm_smem[Kn * FSTR * 2];
    __shared__ int s_j[Kn];
    _Float16 (*feats)[FSTR] = (_Float16 (*)[FSTR])pm_smem;
    float    (*etile)[EDIM + 4] = (float (*)[EDIM + 4])pm_smem;

    int tid = threadIdx.x;
    int bi  = blockIdx.x;                 // (b,i) row
    int b   = bi / Ln;
    int i   = bi - b * Ln;

    if (tid < Kn) s_j[tid] = eidx[(size_t)bi * Kn + tid];
    __syncthreads();

    // --- phase A1: positional features (48 edges x 16 cols = 768 tasks) ---
    int ri_i = ridx[b*Ln + i];
    int ce_i = cenc[b*Ln + i];
    for (int t = tid; t < Kn * 16; t += 256) {
        int e = t >> 4, c = t & 15;
        int j = s_j[e];
        int off  = ri_i - ridx[b*Ln + j];
        int same = (ce_i == cenc[b*Ln + j]);
        int dd = off + MAXREL;
        dd = dd < 0 ? 0 : (dd > 2*MAXREL ? 2*MAXREL : dd);
        int didx = same ? dd : (2*MAXREL + 1);
        feats[e][c] = (_Float16)(posW[didx*16 + c] + posB[c]);
    }
    // --- phase A2: 48 edges x 25 pairs -> 16 RBFs each (1200 tasks) ---
    for (int t = tid; t < Kn * NPAIR; t += 256) {
        int p = t / Kn;
        int e = t - p * Kn;
        int j = s_j[e];
        const float* Ap = atoms5 + ((size_t)(b*Ln + i) * 5 + c_PA[p]) * 3;
        const float* Bp = atoms5 + ((size_t)(b*Ln + j) * 5 + c_PB[p]) * 3;
        float dx = Ap[0]-Bp[0], dy = Ap[1]-Bp[1], dz = Ap[2]-Bp[2];
        float D = sqrtf(dx*dx + dy*dy + dz*dz + 1e-6f);
        #pragma unroll
        for (int r = 0; r < 16; ++r) {
            float u = (D - (2.0f + (20.0f/15.0f) * (float)r)) * 0.8f;  // sigma = 1.25
            feats[e][16 + p*16 + r] = (_Float16)__expf(-u*u);
        }
    }
    __syncthreads();

    // --- phase B: 48(M) x 128(N) x 416(K) GEMM via v_wmma_f32_16x16x32_f16 ---
    int wave = tid >> 5, lane = tid & 31;
    int m = lane & 15, hh = lane >> 4;
    union AV { v16h v; v8h p[2]; };
    union { v8f v; float f[8]; } acc[MT];
    #pragma unroll
    for (int mt = 0; mt < MT; ++mt) acc[mt].v = (v8f){};
    const v16h* wpv = (const v16h*)wp;
    const v8h* rowp[MT];
    #pragma unroll
    for (int mt = 0; mt < MT; ++mt)
        rowp[mt] = (const v8h*)&feats[mt*16 + m][0];     // 16B-aligned (848B row stride)

    // A-fragment double buffer: preload chunk 0
    AV a[2][MT];
    #pragma unroll
    for (int mt = 0; mt < MT; ++mt) {
        a[0][mt].p[0] = rowp[mt][2*hh];        // halfs hh*16 .. +7
        a[0][mt].p[1] = rowp[mt][2*hh + 1];    // halfs hh*16 + 8 .. +7
    }
    #pragma unroll
    for (int kk = 0; kk < NCHUNK; ++kk) {
        int cur = kk & 1, nxt = cur ^ 1;
        v16h bfrag = wpv[(kk * NTILES + wave) * 32 + lane];   // 1 load, 3 WMMAs
        if (kk + 1 < NCHUNK) {
            #pragma unroll
            for (int mt = 0; mt < MT; ++mt) {
                a[nxt][mt].p[0] = rowp[mt][(kk+1)*4 + 2*hh];
                a[nxt][mt].p[1] = rowp[mt][(kk+1)*4 + 2*hh + 1];
            }
        }
        #pragma unroll
        for (int mt = 0; mt < MT; ++mt)
            acc[mt].v = __builtin_amdgcn_wmma_f32_16x16x32_f16(
                false, a[cur][mt].v, false, bfrag, (short)0, acc[mt].v, false, false);
    }
    __syncthreads();   // feats lifetime ends; smem becomes etile

    // D layout: VGPR r -> (row = mt*16 + hh*8 + r, col = wave*16 + (lane&15))
    #pragma unroll
    for (int mt = 0; mt < MT; ++mt)
        #pragma unroll
        for (int r = 0; r < 8; ++r)
            etile[mt*16 + hh*8 + r][wave*16 + m] = acc[mt].f[r];
    __syncthreads();

    // --- phase C: LayerNorm over 128; each wave handles 6 rows ---
    for (int rr = 0; rr < Kn / 8; ++rr) {
        int row = wave * (Kn / 8) + rr;
        float s = 0.f, s2 = 0.f;
        #pragma unroll
        for (int c = lane; c < EDIM; c += 32) { float v = etile[row][c]; s += v; s2 += v*v; }
        #pragma unroll
        for (int o = 16; o > 0; o >>= 1) {
            s  += __shfl_xor(s,  o, 32);
            s2 += __shfl_xor(s2, o, 32);
        }
        float mu   = s  * (1.0f/EDIM);
        float var  = s2 * (1.0f/EDIM) - mu*mu;
        float rstd = rsqrtf(var + 1e-5f);
        float* op = out + ((size_t)bi * Kn + row) * EDIM;
        for (int c = lane; c < EDIM; c += 32)
            op[c] = (etile[row][c] - mu) * rstd * lng[c] + lnb[c];
    }
}

extern "C" void kernel_launch(void* const* d_in, const int* in_sizes, int n_in,
                              void* d_out, int out_size, void* d_ws, size_t ws_size,
                              hipStream_t stream) {
    const float* X    = (const float*)d_in[0];
    const float* mask = (const float*)d_in[1];
    const int*   ridx = (const int*)d_in[2];
    const int*   cenc = (const int*)d_in[3];
    const float* posW = (const float*)d_in[4];
    const float* posB = (const float*)d_in[5];
    const float* edgeW= (const float*)d_in[6];
    const float* lng  = (const float*)d_in[7];
    const float* lnb  = (const float*)d_in[8];
    float* out = (float*)d_out;

    // workspace layout (256B-aligned sections)
    char* ws = (char*)d_ws;
    float*    atoms5 = (float*)ws;                                  // 491,520 B
    int*      eidx   = (int*)(ws + 491520);                         // 1,572,864 B
    _Float16* wp     = (_Float16*)(ws + 491520 + 1572864);          // 106,496 B

    float* eidx_out = out + (size_t)Bn * Ln * Kn * EDIM;            // E_idx (as float) after E

    pm_atoms_kernel<<<(Bn*Ln + 255)/256, 256, 0, stream>>>(X, atoms5);
    pm_wrepack_kernel<<<(NCHUNK*NTILES*32*16 + 255)/256, 256, 0, stream>>>(edgeW, wp);
    pm_topk_kernel<<<Bn*Ln, 256, 0, stream>>>(X, mask, eidx, eidx_out);
    pm_edge_kernel<<<Bn*Ln, 256, 0, stream>>>(
        atoms5, eidx, ridx, cenc, posW, posB, wp, lng, lnb, out);
}
